// DisentangledSelfAttention_49795850830383
// MI455X (gfx1250) — matrix operand order repaired and checked by reading, not compiled, not run on hardware
//
#include <hip/hip_runtime.h>
#include <hip/hip_bf16.h>

// ---------------------------------------------------------------------------
// DeBERTa disentangled self-attention, fused for MI455X (gfx1250, wave32).
// All matmuls go through v_wmma_f32_16x16x32_f16. Attention is flash-style:
// no (B,NH,L,L) tensor is ever materialized in HBM. Working set fits in the
// 192MB L2, so operands stream straight from global (L2 hits) with
// global_prefetch_b8 hints one tile ahead.
// ---------------------------------------------------------------------------

typedef __attribute__((ext_vector_type(16))) _Float16 v16h;  // 8 VGPRs
typedef __attribute__((ext_vector_type(8)))  _Float16 h8;    // 4 VGPRs
typedef __attribute__((ext_vector_type(8)))  float    v8f;   // 8 VGPRs

union AFrag { v16h v; h8 h[2]; };

#define WMMA_F16(A, B, C) \
    __builtin_amdgcn_wmma_f32_16x16x32_f16(false, (A), false, (B), (short)0, (C), false, false)

static constexpr int   kB_ = 4, kL = 1024, kH = 1024, kNH = 16, kD = 64;
static constexpr int   kSpan2 = 1024;                 // 2 * att_span
static constexpr float kInvSqrt = 0.07216878364870323f; // 1/sqrt(64*3)

// ---------------------------------------------------------------------------
// f32 -> f16 elementwise convert
// ---------------------------------------------------------------------------
__global__ void cvt_f16_kernel(const float* __restrict__ in, _Float16* __restrict__ out, int n) {
    int i = blockIdx.x * blockDim.x + threadIdx.x;
    if (i < n) out[i] = (_Float16)in[i];
}

// ---------------------------------------------------------------------------
// Transpose K x N f32 -> N x K f16 (LDS-tiled; K,N multiples of 32)
// ---------------------------------------------------------------------------
__global__ __launch_bounds__(256)
void transpose_f16_kernel(const float* __restrict__ in, _Float16* __restrict__ out, int K, int N) {
    __shared__ float tile[32][33];
    int n0 = blockIdx.x * 32, k0 = blockIdx.y * 32;
    #pragma unroll
    for (int yy = threadIdx.y; yy < 32; yy += 8)
        tile[yy][threadIdx.x] = in[(size_t)(k0 + yy) * N + n0 + threadIdx.x];
    __syncthreads();
    #pragma unroll
    for (int yy = threadIdx.y; yy < 32; yy += 8)
        out[(size_t)(n0 + yy) * K + k0 + threadIdx.x] = (_Float16)tile[threadIdx.x][yy];
}

// ---------------------------------------------------------------------------
// Generic WMMA GEMM: C(MxN,f16) = scale * (A(MxK,f16) @ BT(NxK,f16)^T + bias)
// One wave computes a 16x64 output tile. blockDim = 128 (4 waves).
// ---------------------------------------------------------------------------
__global__ __launch_bounds__(128)
void gemm16x64_kernel(const _Float16* __restrict__ A, const _Float16* __restrict__ BT,
                      const float* __restrict__ bias, float scale,
                      _Float16* __restrict__ C, int N, int K, int tiles_m) {
    int wid  = blockIdx.x * 4 + (threadIdx.x >> 5);
    int tm   = wid % tiles_m;
    int tn   = wid / tiles_m;
    int lane = threadIdx.x & 31;
    int n16  = lane & 15;
    int hi   = lane >> 4;

    v8f acc[4] = {v8f{}, v8f{}, v8f{}, v8f{}};
    const _Float16* arow = A + (size_t)(tm * 16 + n16) * K;

    for (int k0 = 0; k0 < K; k0 += 32) {
        if (k0 + 128 < K) {                       // one 256B stretch ahead
            __builtin_prefetch(arow + k0 + 128, 0, 1);
            #pragma unroll
            for (int t = 0; t < 4; ++t)
                __builtin_prefetch(BT + (size_t)(tn * 64 + t * 16 + n16) * K + k0 + 128, 0, 1);
        }
        AFrag a;
        a.h[0] = *(const h8*)(arow + k0 + hi * 16);
        a.h[1] = *(const h8*)(arow + k0 + hi * 16 + 8);
        #pragma unroll
        for (int t = 0; t < 4; ++t) {
            int ncol = tn * 64 + t * 16 + n16;
            v16h bf  = *(const v16h*)(BT + (size_t)ncol * K + k0 + hi * 16);
            acc[t]   = WMMA_F16(a.v, bf, acc[t]);
        }
    }
    #pragma unroll
    for (int t = 0; t < 4; ++t) {
        int   ncol = tn * 64 + t * 16 + n16;
        float bv   = bias ? bias[ncol] : 0.0f;
        #pragma unroll
        for (int r = 0; r < 8; ++r) {
            int orow = tm * 16 + r + 8 * hi;
            C[(size_t)orow * N + ncol] = (_Float16)((acc[t][r] + bv) * scale);
        }
    }
}

// ---------------------------------------------------------------------------
// QKV GEMM with fused epilogue: split into q (bias + 1/sqrt(192) scale),
// k (plain), vT (bias, stored d-major per head for PV B-fragments).
// M = B*L = 4096, N = 3072, K = 1024.
// ---------------------------------------------------------------------------
__global__ __launch_bounds__(128)
void gemm_qkv_kernel(const _Float16* __restrict__ A, const _Float16* __restrict__ BT,
                     const float* __restrict__ qb, const float* __restrict__ vb,
                     _Float16* __restrict__ q, _Float16* __restrict__ k,
                     _Float16* __restrict__ vT) {
    int wid  = blockIdx.x * 4 + (threadIdx.x >> 5);
    int tm   = wid & 255;      // 256 row tiles
    int tn   = wid >> 8;       // 48 column tiles of 64
    int lane = threadIdx.x & 31;
    int n16  = lane & 15;
    int hi   = lane >> 4;

    v8f acc[4] = {v8f{}, v8f{}, v8f{}, v8f{}};
    const _Float16* arow = A + (size_t)(tm * 16 + n16) * kH;

    for (int k0 = 0; k0 < kH; k0 += 32) {
        if (k0 + 128 < kH) {
            __builtin_prefetch(arow + k0 + 128, 0, 1);
            #pragma unroll
            for (int t = 0; t < 4; ++t)
                __builtin_prefetch(BT + (size_t)(tn * 64 + t * 16 + n16) * kH + k0 + 128, 0, 1);
        }
        AFrag a;
        a.h[0] = *(const h8*)(arow + k0 + hi * 16);
        a.h[1] = *(const h8*)(arow + k0 + hi * 16 + 8);
        #pragma unroll
        for (int t = 0; t < 4; ++t) {
            int ncol = tn * 64 + t * 16 + n16;
            v16h bf  = *(const v16h*)(BT + (size_t)ncol * kH + k0 + hi * 16);
            acc[t]   = WMMA_F16(a.v, bf, acc[t]);
        }
    }
    #pragma unroll
    for (int t = 0; t < 4; ++t) {
        int ncol = tn * 64 + t * 16 + n16;   // 0..3071
        int sec  = ncol >> 10;               // 0=q 1=k 2=v
        int w    = ncol & 1023;              // (head, d) packed
        #pragma unroll
        for (int r = 0; r < 8; ++r) {
            int   orow = tm * 16 + r + 8 * hi;   // 0..4095 == b*1024+i
            float val  = acc[t][r];
            if (sec == 0) {
                q[(size_t)orow * kH + w] = (_Float16)((val + qb[w]) * kInvSqrt);
            } else if (sec == 1) {
                k[(size_t)orow * kH + w] = (_Float16)val;
            } else {
                int b = orow >> 10, i = orow & 1023;
                // vT layout: [b][h][d][j]
                vT[(((size_t)(b * kNH) << 6) + w) * (size_t)kL + i] =
                    (_Float16)(val + vb[w]);
            }
        }
    }
}

// ---------------------------------------------------------------------------
// Fused disentangled attention. One wave per (b, h, 16-row tile).
// scores[i,j] = q_i.k_j + q_i.P[d] + k_j.Q[d],  d = clip(i-j+512, 0, 1023).
// Online softmax, PV accumulation; all matmuls via WMMA.
// ---------------------------------------------------------------------------
__global__ __launch_bounds__(32)
void attn_kernel(const _Float16* __restrict__ q, const _Float16* __restrict__ k,
                 const _Float16* __restrict__ vT, const _Float16* __restrict__ P,
                 const _Float16* __restrict__ Q, const unsigned char* __restrict__ mask,
                 float* __restrict__ out) {
    int wg = blockIdx.x;         // B*NH*(L/16) = 4096
    int it = wg & 63;
    int h  = (wg >> 6) & 15;
    int b  = wg >> 10;
    int i0 = it * 16;

    int lane = threadIdx.x;
    int n16  = lane & 15;
    int hi   = lane >> 4;

    __shared__ __align__(64) float    ct_lds[16][32];
    __shared__ __align__(64) float    pt_lds[16][32];
    __shared__ __align__(64) _Float16 p_lds[16][32];

    // Persistent q A-fragments (16 rows x 64 K, two K-chunks)
    const _Float16* qrow = q + (size_t)(b * kL + i0 + n16) * kH + h * kD;
    AFrag qA[2];
    #pragma unroll
    for (int c = 0; c < 2; ++c) {
        qA[c].h[0] = *(const h8*)(qrow + c * 32 + hi * 16);
        qA[c].h[1] = *(const h8*)(qrow + c * 32 + hi * 16 + 8);
    }

    v8f ctx[4] = {v8f{}, v8f{}, v8f{}, v8f{}};
    float mrun[8], lrun[8];
    #pragma unroll
    for (int r = 0; r < 8; ++r) { mrun[r] = -1e30f; lrun[r] = 0.0f; }

    const _Float16* kbase = k + (size_t)(b * kL) * kH + h * kD;
    const _Float16* Pb    = P + h * kD;
    const _Float16* Qb    = Q + h * kD;
    const _Float16* vTb   = vT + ((size_t)(b * kNH + h) << 6) * (size_t)kL;
    const unsigned char* mb = mask + (size_t)b * kL * kL;

    for (int j0 = 0; j0 < kL; j0 += 32) {
        // Prefetch next j-tile's k rows and vT rows while this tile computes.
        if (j0 + 32 < kL) {
            __builtin_prefetch(kbase + (size_t)(j0 + 32 + n16) * kH, 0, 1);
            __builtin_prefetch(kbase + (size_t)(j0 + 48 + n16) * kH, 0, 1);
            __builtin_prefetch(vTb + (size_t)(n16) * kL + j0 + 32, 0, 1);
            __builtin_prefetch(vTb + (size_t)(n16 + 16) * kL + j0 + 32, 0, 1);
            __builtin_prefetch(vTb + (size_t)(n16 + 32) * kL + j0 + 32, 0, 1);
            __builtin_prefetch(vTb + (size_t)(n16 + 48) * kL + j0 + 32, 0, 1);
        }

        float sreg[2][8];
        #pragma unroll
        for (int js = 0; js < 2; ++js) {
            int jt = j0 + js * 16;
            const _Float16* kr = kbase + (size_t)(jt + n16) * kH;
            // k as B-fragment (for q.k^T) and A-fragment (for p2c)
            v16h  kBf[2];
            AFrag kAf[2];
            #pragma unroll
            for (int c = 0; c < 2; ++c) {
                kBf[c]      = *(const v16h*)(kr + c * 32 + hi * 16);
                kAf[c].h[0] = *(const h8*)(kr + c * 32 + hi * 16);
                kAf[c].h[1] = *(const h8*)(kr + c * 32 + hi * 16 + 8);
            }
            // content-content scores
            v8f sc = {};
            sc = WMMA_F16(qA[0].v, kBf[0], sc);
            sc = WMMA_F16(qA[1].v, kBf[1], sc);

            // relative-position band: all needed deltas live in [s0, s0+31]
            int s0 = i0 - jt + 497;
            s0 = s0 < 0 ? 0 : (s0 > 992 ? 992 : s0);

            v8f ct[2], pt[2];
            #pragma unroll
            for (int ss = 0; ss < 2; ++ss) {
                const _Float16* pr = Pb + (size_t)(s0 + ss * 16 + n16) * kH;
                v16h pB0 = *(const v16h*)(pr + hi * 16);
                v16h pB1 = *(const v16h*)(pr + 32 + hi * 16);
                v8f a = {};
                a = WMMA_F16(qA[0].v, pB0, a);
                a = WMMA_F16(qA[1].v, pB1, a);
                ct[ss] = a;

                const _Float16* qr2 = Qb + (size_t)(s0 + ss * 16 + n16) * kH;
                v16h qB0 = *(const v16h*)(qr2 + hi * 16);
                v16h qB1 = *(const v16h*)(qr2 + 32 + hi * 16);
                v8f c2 = {};
                c2 = WMMA_F16(kAf[0].v, qB0, c2);
                c2 = WMMA_F16(kAf[1].v, qB1, c2);
                pt[ss] = c2;
            }
            // spill band results to LDS for the per-element diagonal gather
            #pragma unroll
            for (int r = 0; r < 8; ++r) {
                int m = r + 8 * hi;
                ct_lds[m][n16]      = ct[0][r];
                ct_lds[m][16 + n16] = ct[1][r];
                pt_lds[m][n16]      = pt[0][r];
                pt_lds[m][16 + n16] = pt[1][r];
            }
            asm volatile("s_wait_dscnt 0" ::: "memory");
            #pragma unroll
            for (int r = 0; r < 8; ++r) {
                int m = r + 8 * hi;
                int i = i0 + m;
                int j = jt + n16;
                int delta = i - j + 512;
                delta = delta < 0 ? 0 : (delta > 1023 ? 1023 : delta);
                int idx = delta - s0;
                float sval = sc[r] + ct_lds[m][idx] + pt_lds[n16][idx];
                sval = mb[(size_t)i * kL + j] ? sval : -3.0e38f;
                sreg[js][r] = sval;
            }
        }

        // ---- online softmax over the 16x32 tile (rows span 16-lane halves) ----
        #pragma unroll
        for (int r = 0; r < 8; ++r) {
            float t = fmaxf(sreg[0][r], sreg[1][r]);
            t = fmaxf(t, __shfl_xor(t, 1, 32));
            t = fmaxf(t, __shfl_xor(t, 2, 32));
            t = fmaxf(t, __shfl_xor(t, 4, 32));
            t = fmaxf(t, __shfl_xor(t, 8, 32));
            float mnew = fmaxf(mrun[r], t);
            float scl  = __expf(mrun[r] - mnew);
            mrun[r] = mnew;
            lrun[r] *= scl;
            ctx[0][r] *= scl; ctx[1][r] *= scl; ctx[2][r] *= scl; ctx[3][r] *= scl;

            float p0 = __expf(sreg[0][r] - mnew);
            float p1 = __expf(sreg[1][r] - mnew);
            float ps = p0 + p1;
            ps += __shfl_xor(ps, 1, 32);
            ps += __shfl_xor(ps, 2, 32);
            ps += __shfl_xor(ps, 4, 32);
            ps += __shfl_xor(ps, 8, 32);
            lrun[r] += ps;

            int m = r + 8 * hi;
            p_lds[m][n16]      = (_Float16)p0;
            p_lds[m][16 + n16] = (_Float16)p1;
        }
        asm volatile("s_wait_dscnt 0" ::: "memory");

        // probs re-read from LDS in A-fragment layout (16 x 32)
        AFrag pA;
        pA.h[0] = *(const h8*)&p_lds[n16][hi * 16];
        pA.h[1] = *(const h8*)&p_lds[n16][hi * 16 + 8];

        // PV: ctx(16x64) += probs(16x32) @ v(32x64); vT gives contiguous B-frags
        #pragma unroll
        for (int dsub = 0; dsub < 4; ++dsub) {
            v16h vB = *(const v16h*)(vTb + (size_t)(dsub * 16 + n16) * kL + j0 + hi * 16);
            ctx[dsub] = WMMA_F16(pA.v, vB, ctx[dsub]);
        }
    }

    // ---- epilogue: normalize and store (B, L, H) f32 ----
    float* ob = out + (size_t)(b * kL + i0) * kH + h * kD;
    #pragma unroll
    for (int r = 0; r < 8; ++r) {
        float inv = 1.0f / lrun[r];
        int m = r + 8 * hi;
        #pragma unroll
        for (int dsub = 0; dsub < 4; ++dsub)
            ob[(size_t)m * kH + dsub * 16 + n16] = ctx[dsub][r] * inv;
    }
}

// ---------------------------------------------------------------------------
// Host launcher
// ---------------------------------------------------------------------------
extern "C" void kernel_launch(void* const* d_in, const int* in_sizes, int n_in,
                              void* d_out, int out_size, void* d_ws, size_t ws_size,
                              hipStream_t stream) {
    (void)in_sizes; (void)n_in; (void)out_size; (void)ws_size;

    const float*         hidden  = (const float*)d_in[0];
    const unsigned char* mask    = (const unsigned char*)d_in[1];
    const float*         rel     = (const float*)d_in[2];
    const float*         W_in    = (const float*)d_in[3];
    const float*         q_bias  = (const float*)d_in[4];
    const float*         v_bias  = (const float*)d_in[5];
    const float*         W_pos   = (const float*)d_in[6];
    const float*         W_posq  = (const float*)d_in[7];
    const float*         b_posq  = (const float*)d_in[8];
    float*               out     = (float*)d_out;

    const size_t BL = (size_t)kB_ * kL;      // 4096
    char*  ws  = (char*)d_ws;
    size_t off = 0;
    auto carve = [&](size_t bytes) -> char* {
        char* p = ws + off;
        off += (bytes + 255) & ~(size_t)255;
        return p;
    };
    _Float16* hiddenH = (_Float16*)carve(BL * kH * 2);            // 8 MB
    _Float16* relH    = (_Float16*)carve((size_t)kSpan2 * kH * 2);// 2 MB
    _Float16* WinT    = (_Float16*)carve((size_t)3 * kH * kH * 2);// 6 MB
    _Float16* WposT   = (_Float16*)carve((size_t)kH * kH * 2);    // 2 MB
    _Float16* WposqT  = (_Float16*)carve((size_t)kH * kH * 2);    // 2 MB
    _Float16* qbuf    = (_Float16*)carve(BL * kH * 2);            // 8 MB
    _Float16* kbuf    = (_Float16*)carve(BL * kH * 2);            // 8 MB
    _Float16* vTbuf   = (_Float16*)carve(BL * kH * 2);            // 8 MB
    _Float16* Pbuf    = (_Float16*)carve((size_t)kSpan2 * kH * 2);// 2 MB
    _Float16* Qbuf    = (_Float16*)carve((size_t)kSpan2 * kH * 2);// 2 MB

    // 1) convert activations to f16
    int nh_ = (int)(BL * kH);
    cvt_f16_kernel<<<(nh_ + 255) / 256, 256, 0, stream>>>(hidden, hiddenH, nh_);
    int nr_ = kSpan2 * kH;
    cvt_f16_kernel<<<(nr_ + 255) / 256, 256, 0, stream>>>(rel, relH, nr_);

    // 2) transpose + convert weights (B-fragment friendly N x K layout)
    dim3 tb(32, 8);
    transpose_f16_kernel<<<dim3(96, 32), tb, 0, stream>>>(W_in,   WinT,   kH, 3 * kH);
    transpose_f16_kernel<<<dim3(32, 32), tb, 0, stream>>>(W_pos,  WposT,  kH, kH);
    transpose_f16_kernel<<<dim3(32, 32), tb, 0, stream>>>(W_posq, WposqT, kH, kH);

    // 3) QKV projection (fused bias/scale/split/transpose epilogue)
    //    256 row-tiles x 48 col-tiles = 12288 waves -> 3072 blocks of 128
    gemm_qkv_kernel<<<3072, 128, 0, stream>>>(hiddenH, WinT, q_bias, v_bias,
                                              qbuf, kbuf, vTbuf);

    // 4) positional projections: P = rel@W_pos ; Q = (rel@W_posq + b)/sqrt(192)
    //    64 x 16 tiles = 1024 waves -> 256 blocks
    gemm16x64_kernel<<<256, 128, 0, stream>>>(relH, WposT, nullptr, 1.0f,
                                              Pbuf, kH, kH, 64);
    gemm16x64_kernel<<<256, 128, 0, stream>>>(relH, WposqT, b_posq, kInvSqrt,
                                              Qbuf, kH, kH, 64);

    // 5) fused attention: one wave per (b, h, 16-row tile)
    attn_kernel<<<kB_ * kNH * (kL / 16), 32, 0, stream>>>(
        qbuf, kbuf, vTbuf, Pbuf, Qbuf, mask, out);
}
